// MultHeadAttentionLayer_45724221833755
// MI455X (gfx1250) — matrix operand order repaired and checked by reading, compiled, not run on hardware
//
#include <hip/hip_runtime.h>
#include <hip/hip_bf16.h>

typedef __attribute__((ext_vector_type(16))) _Float16 v16h;
typedef __attribute__((ext_vector_type(8)))  _Float16 v8h;
typedef __attribute__((ext_vector_type(8)))  float    v8f;

constexpr int NB = 1024, NF = 128, NDIN = 64, NH = 8, ND = 64;

// f16 LDS strides (in halfs): 144B / 272B rows -> 16B aligned chunks everywhere
constexpr int XS = 72;    // for 64-wide matrices
constexpr int TS = 136;   // for 128-wide matrices

// LDS layout (offsets in halfs)
constexpr int OFF_XH  = 0;                 // X      [128][64] row-major (A-operand)
constexpr int OFF_WQT = OFF_XH  + NF*XS;   // Wq^T   [64][64]  ([d][i], B-operand)
constexpr int OFF_WKT = OFF_WQT + ND*XS;   // Wk^T   [64][64]
constexpr int OFF_WRT = OFF_WKT + ND*XS;   // Wr^T   [64][64]
constexpr int OFF_QM  = OFF_WRT + ND*XS;   // Q      [128][64] row-major (A for scores)
constexpr int OFF_KM  = OFF_QM  + NF*XS;   // K=V    [128][64] row-major (== B^T for Q.K^T)
constexpr int OFF_KTM = OFF_KM  + NF*XS;   // V^T    [64][128] ([d][g], B-operand for A.V)
constexpr int OFF_AM  = OFF_KTM + ND*TS;   // sigmoid(S) [128][128] (A for A.V)
constexpr int SMEM_HALFS = OFF_AM + NF*TS;
constexpr size_t SMEM_BYTES = (size_t)SMEM_HALFS * sizeof(_Float16);
// f32 O-buffer [128][72] (36864B) overlays OFF_XH..OFF_QM (46080B), dead by then.

__device__ __forceinline__ void wmma_f16(v16h a, v16h b, v8f& c) {
  c = __builtin_amdgcn_wmma_f32_16x16x32_f16(false, a, false, b, (short)0, c,
                                             false, false);
}

// A-fragment: 16x32 f16 tile at (row0, k0) from a row-major LDS matrix.
// lane L<16: row=L, K in {k0..k0+7} u {k0+16..k0+23}; lane>=16: K ranges +8.
__device__ __forceinline__ v16h load_a_frag(const _Float16* base, int stride,
                                            int row0, int k0) {
  int lane = threadIdx.x & 31;
  int hh = lane >> 4;
  const _Float16* p = base + (row0 + (lane & 15)) * stride + k0 + hh * 8;
  v8h lo = *(const v8h*)p;
  v8h hi = *(const v8h*)(p + 16);
  return __builtin_shufflevector(lo, hi, 0,1,2,3,4,5,6,7,8,9,10,11,12,13,14,15);
}

// B-fragment: 32x16 f16 tile B[k][n] where baseT holds B^T ([n][k] row-major).
// lane L<16: col n=L, K in {k0..k0+7} u {k0+16..k0+23}; lane>=16: K ranges +8.
__device__ __forceinline__ v16h load_bt_frag(const _Float16* baseT, int stride,
                                             int k0, int n0) {
  int lane = threadIdx.x & 31;
  int hh = lane >> 4;
  const _Float16* p = baseT + (n0 + (lane & 15)) * stride + k0 + hh * 8;
  v8h lo = *(const v8h*)p;
  v8h hi = *(const v8h*)(p + 16);
  return __builtin_shufflevector(lo, hi, 0,1,2,3,4,5,6,7,8,9,10,11,12,13,14,15);
}

// C/D fragment store: VGPR i -> row row0 + (lane/16)*8 + i, col col0 + lane%16
__device__ __forceinline__ void store_c_f16(_Float16* base, int stride,
                                            int row0, int col0, v8f c) {
  int lane = threadIdx.x & 31;
  int hh = lane >> 4;
  int n = col0 + (lane & 15);
#pragma unroll
  for (int i = 0; i < 8; ++i)
    base[(row0 + hh * 8 + i) * stride + n] = (_Float16)c[i];
}

__global__ void
__launch_bounds__(256, 2)
mha_sigmoid_kernel(const float* __restrict__ x,
                   const float* __restrict__ qw,
                   const float* __restrict__ kw,
                   const float* __restrict__ rw,
                   const float* __restrict__ gamma,
                   const float* __restrict__ beta,
                   float* __restrict__ out) {
  extern __shared__ __align__(16) _Float16 smem[];
  _Float16* xh  = smem + OFF_XH;
  _Float16* wqT = smem + OFF_WQT;
  _Float16* wkT = smem + OFF_WKT;
  _Float16* wrT = smem + OFF_WRT;
  _Float16* qm  = smem + OFF_QM;
  _Float16* km  = smem + OFF_KM;
  _Float16* ktm = smem + OFF_KTM;
  _Float16* am  = smem + OFF_AM;
  float*    om  = (float*)smem;  // overlays X/W region (dead after phase 1)

  const int h = blockIdx.x & (NH - 1);
  const int b = blockIdx.x >> 3;
  const int tid = threadIdx.x;
  const int lane = tid & 31;
  const int hh = lane >> 4;
  const int ln15 = lane & 15;
  const int rt = tid >> 5;  // wave id == row-tile 0..7

  // ---------------- Phase 0: stage X (f16) and W^T (f16) into LDS ----------
  {
    const float* xb = x + (size_t)b * NF * NDIN;
#pragma unroll 4
    for (int idx = tid; idx < NF * NDIN; idx += 256) {
      int f = idx >> 6, i = idx & 63;
      xh[f * XS + i] = (_Float16)xb[idx];
    }
    const size_t wstep = (size_t)NH * ND;  // row stride of [DIN,H,D]
#pragma unroll 4
    for (int idx = tid; idx < NDIN * ND; idx += 256) {
      int i = idx >> 6, d = idx & 63;
      size_t g = (size_t)i * wstep + (size_t)h * ND + d;
      wqT[d * XS + i] = (_Float16)qw[g];
      wkT[d * XS + i] = (_Float16)kw[g];
      wrT[d * XS + i] = (_Float16)rw[g];
    }
  }
  __syncthreads();

  // ---------------- Phase 1: Q = X Wq, K = V = X Wk, R = X Wr --------------
  const size_t hb_base = ((size_t)h * NB + b) * NF * ND;
  const size_t res_base = (size_t)NH * NB * NF * ND + hb_base;
  {
    v16h ax0 = load_a_frag(xh, XS, rt * 16, 0);
    v16h ax1 = load_a_frag(xh, XS, rt * 16, 32);
#pragma unroll
    for (int ct = 0; ct < 4; ++ct) {
      v8f cq = {}, ck = {}, cr = {};
      wmma_f16(ax0, load_bt_frag(wqT, XS, 0,  ct * 16), cq);
      wmma_f16(ax1, load_bt_frag(wqT, XS, 32, ct * 16), cq);
      wmma_f16(ax0, load_bt_frag(wkT, XS, 0,  ct * 16), ck);
      wmma_f16(ax1, load_bt_frag(wkT, XS, 32, ct * 16), ck);
      wmma_f16(ax0, load_bt_frag(wrT, XS, 0,  ct * 16), cr);
      wmma_f16(ax1, load_bt_frag(wrT, XS, 32, ct * 16), cr);
      store_c_f16(qm, XS, rt * 16, ct * 16, cq);
      store_c_f16(km, XS, rt * 16, ct * 16, ck);
      // transposed copy of K (=V) for the A.V B-operand: ktm[d][g].
      // Per lane the 8 values are contiguous halfs -> one ds_store_b128.
      {
        int n = ct * 16 + ln15;
        v8h pk;
#pragma unroll
        for (int i = 0; i < 8; ++i) pk[i] = (_Float16)ck[i];
        *(v8h*)(ktm + n * TS + rt * 16 + hh * 8) = pk;
      }
      // res = X Wr straight to global (f32, no layernorm)
      {
        int n = ct * 16 + ln15;
#pragma unroll
        for (int i = 0; i < 8; ++i) {
          int r = rt * 16 + hh * 8 + i;
          out[res_base + (size_t)r * ND + n] = cr[i];
        }
      }
    }
  }
  __syncthreads();

  // ---------------- Phase 2: A = sigmoid(Q K^T / 8) ------------------------
  {
    v16h aq0 = load_a_frag(qm, XS, rt * 16, 0);
    v16h aq1 = load_a_frag(qm, XS, rt * 16, 32);
#pragma unroll
    for (int gt = 0; gt < 8; ++gt) {
      v8f c = {};
      wmma_f16(aq0, load_bt_frag(km, XS, 0,  gt * 16), c);
      wmma_f16(aq1, load_bt_frag(km, XS, 32, gt * 16), c);
      int n = gt * 16 + ln15;
#pragma unroll
      for (int i = 0; i < 8; ++i) {
        float s = c[i] * 0.125f;                  // 1/sqrt(64)
        float a = 1.0f / (1.0f + __expf(-s));     // sigmoid, not softmax
        am[(rt * 16 + hh * 8 + i) * TS + n] = (_Float16)a;
      }
    }
  }
  __syncthreads();

  // ---------------- Phase 3: O = A V  (f32 into LDS) -----------------------
  {
    v16h aa0 = load_a_frag(am, TS, rt * 16, 0);
    v16h aa1 = load_a_frag(am, TS, rt * 16, 32);
    v16h aa2 = load_a_frag(am, TS, rt * 16, 64);
    v16h aa3 = load_a_frag(am, TS, rt * 16, 96);
#pragma unroll
    for (int ct = 0; ct < 4; ++ct) {
      v8f c = {};
      wmma_f16(aa0, load_bt_frag(ktm, TS, 0,  ct * 16), c);
      wmma_f16(aa1, load_bt_frag(ktm, TS, 32, ct * 16), c);
      wmma_f16(aa2, load_bt_frag(ktm, TS, 64, ct * 16), c);
      wmma_f16(aa3, load_bt_frag(ktm, TS, 96, ct * 16), c);
      int n = ct * 16 + ln15;
#pragma unroll
      for (int i = 0; i < 8; ++i)
        om[(rt * 16 + hh * 8 + i) * XS + n] = c[i];
    }
  }
  __syncthreads();

  // ------- Phase 4: LayerNorm over D=64; 2 threads/row, float4 I/O ---------
  {
    int r  = tid >> 1;      // row 0..127
    int hf = tid & 1;       // half-row: cols hf*32 .. hf*32+31 (partner = lane^1)
    const float4* rowv = (const float4*)(om + r * XS + hf * 32);
    float4 vbuf[8];
    float s = 0.0f;
#pragma unroll
    for (int j = 0; j < 8; ++j) {
      vbuf[j] = rowv[j];
      s += vbuf[j].x + vbuf[j].y + vbuf[j].z + vbuf[j].w;
    }
    s += __shfl_xor(s, 1, 32);
    float mu = s * (1.0f / ND);
    float vs = 0.0f;
#pragma unroll
    for (int j = 0; j < 8; ++j) {
      float tx = vbuf[j].x - mu, ty = vbuf[j].y - mu;
      float tz = vbuf[j].z - mu, tw = vbuf[j].w - mu;
      vs += tx * tx + ty * ty + tz * tz + tw * tw;
    }
    vs += __shfl_xor(vs, 1, 32);
    float inv = rsqrtf(vs * (1.0f / ND) + 1e-3f);

    const float4* gv = (const float4*)(gamma + hf * 32);
    const float4* bv = (const float4*)(beta + hf * 32);
    float4* outv = (float4*)(out + hb_base + (size_t)r * ND + hf * 32);
#pragma unroll
    for (int j = 0; j < 8; ++j) {
      float4 g = gv[j], bb = bv[j], o;
      o.x = (vbuf[j].x - mu) * inv * g.x + bb.x;
      o.y = (vbuf[j].y - mu) * inv * g.y + bb.y;
      o.z = (vbuf[j].z - mu) * inv * g.z + bb.z;
      o.w = (vbuf[j].w - mu) * inv * g.w + bb.w;
      outv[j] = o;
    }
  }
}

extern "C" void kernel_launch(void* const* d_in, const int* in_sizes, int n_in,
                              void* d_out, int out_size, void* d_ws, size_t ws_size,
                              hipStream_t stream) {
  const float* x     = (const float*)d_in[0];
  const float* qw    = (const float*)d_in[1];
  const float* kw    = (const float*)d_in[2];
  const float* rw    = (const float*)d_in[3];
  const float* gamma = (const float*)d_in[4];
  const float* beta  = (const float*)d_in[5];
  float* out = (float*)d_out;

  dim3 grid(NH * NB);  // one workgroup per (head, batch)
  mha_sigmoid_kernel<<<grid, 256, SMEM_BYTES, stream>>>(x, qw, kw, rw, gamma,
                                                        beta, out);
}
